// CongestionGAT_16870631539033
// MI455X (gfx1250) — matrix operand order repaired and checked by reading, not compiled, not run on hardware
//
#include <hip/hip_runtime.h>
#include <math.h>

typedef __attribute__((ext_vector_type(16))) _Float16 v16h;
typedef __attribute__((ext_vector_type(8)))  float    v8f;

#define DEVINL __device__ __forceinline__

DEVINL float lrelu02(float v) { return v > 0.f ? v : 0.2f * v; }
DEVINL float eluf(float v)    { return v > 0.f ? v : (__expf(v) - 1.f); }

// order-preserving float<->uint encoding for atomicMax-based segment max
DEVINL unsigned fenc(float f) {
  unsigned u = __float_as_uint(f);
  return (u & 0x80000000u) ? ~u : (u | 0x80000000u);
}
DEVINL float fdec(unsigned u) {
  return (u & 0x80000000u) ? __uint_as_float(u & 0x7fffffffu) : __uint_as_float(~u);
}

// edge e of [real edges | self loops]
DEVINL void get_edge(const int* ei, int E, int e, int& s, int& d) {
  if (e < E) { s = ei[e]; d = ei[E + e]; }
  else       { s = e - E; d = e - E; }
}

__global__ void k_zero(float* __restrict__ p, size_t n) {
  size_t i = (size_t)blockIdx.x * blockDim.x + threadIdx.x;
  size_t stride = (size_t)gridDim.x * blockDim.x;
  for (; i < n; i += stride) p[i] = 0.f;
}

// Pack W[K,M] (row-major f32) into per-lane B fragments (f16), zero-padded in K.
// Bp layout: for k-tile t, column c, half-group g: 16 contiguous halfs at
//   (((t*M + c)*2 + g)*16 + e)  holding W[t*32 + (e<8 ? g*8+e : 16+g*8+(e-8)), c]
template<int K, int M>
__global__ void k_packB(const float* __restrict__ W, _Float16* __restrict__ Bp) {
  const int TILES = (K + 31) / 32;
  int t = blockIdx.x * blockDim.x + threadIdx.x;
  if (t >= TILES * M * 32) return;
  int e  = t & 15;
  int g  = (t >> 4) & 1;
  int c  = (t >> 5) % M;
  int tl = t / (M * 32);
  int k  = tl * 32 + (e < 8 ? g * 8 + e : 16 + g * 8 + (e - 8));
  Bp[t] = (k < K) ? (_Float16)W[(size_t)k * M + c] : (_Float16)0.f;
}

// OUT[N,M] = X[N,K] @ W[K,M]. One wave per 16x16 tile, M/16 waves per block
// covering a 16-row strip. Requires N % 16 == 0 (true: N = 100000).
// B comes pre-packed in fragment order (f16), A loads as float4 + cvt.
template<int K, int M>
__global__ void k_gemm_wmma(const float* __restrict__ X, const _Float16* __restrict__ Bp,
                            float* __restrict__ OUT) {
  const int lane = threadIdx.x & 31;
  const int wave = threadIdx.x >> 5;
  const int row0 = blockIdx.x * 16;
  const int mrow = lane & 15;
  const int hi   = lane >> 4;            // 0 or 1
  const int ncol = wave * 16 + (lane & 15);

  const float* xrow = X + (size_t)(row0 + mrow) * K + hi * 8;
  const _Float16* bfrag = Bp + (((size_t)ncol * 2) + hi) * 16;

  v8f acc = {};
#pragma unroll
  for (int k0 = 0; k0 < K; k0 += 32) {
    v16h a;
    // elems 0..7 -> K = k0 + hi*8 + (0..7)   (always in range: K % 16 == 0)
    float4 p0 = *(const float4*)(xrow + k0);
    float4 p1 = *(const float4*)(xrow + k0 + 4);
    a[0] = (_Float16)p0.x; a[1] = (_Float16)p0.y; a[2] = (_Float16)p0.z; a[3] = (_Float16)p0.w;
    a[4] = (_Float16)p1.x; a[5] = (_Float16)p1.y; a[6] = (_Float16)p1.z; a[7] = (_Float16)p1.w;
    // elems 8..15 -> K = k0 + 16 + hi*8 + (0..7)
    if (k0 + 32 <= K) {
      float4 p2 = *(const float4*)(xrow + k0 + 16);
      float4 p3 = *(const float4*)(xrow + k0 + 20);
      a[8]  = (_Float16)p2.x; a[9]  = (_Float16)p2.y; a[10] = (_Float16)p2.z; a[11] = (_Float16)p2.w;
      a[12] = (_Float16)p3.x; a[13] = (_Float16)p3.y; a[14] = (_Float16)p3.z; a[15] = (_Float16)p3.w;
    } else {
#pragma unroll
      for (int j = 8; j < 16; ++j) a[j] = (_Float16)0.f;
    }
    // one contiguous 32B fragment per lane per k-tile
    v16h b = *(const v16h*)(bfrag + (size_t)(k0 / 32) * M * 32);
    acc = __builtin_amdgcn_wmma_f32_16x16x32_f16(false, a, false, b,
                                                 (short)0, acc, false, false);
  }

#pragma unroll
  for (int r = 0; r < 8; ++r) {          // C/D: VGPR r -> row r (+8 for upper lanes)
    OUT[(size_t)(row0 + r + hi * 8) * M + ncol] = acc[r];
  }
}

// a_src[n,h] = sum_c H[n,h,c]*wS[h,c] ; same for a_dst
template<int NH>
__global__ void k_attn(const float* __restrict__ Hf, const float* __restrict__ wS,
                       const float* __restrict__ wD, float* __restrict__ aS,
                       float* __restrict__ aD, int Nn) {
  int t = blockIdx.x * blockDim.x + threadIdx.x;
  if (t >= Nn * NH) return;
  int n = t / NH, h = t - n * NH;
  const float* row = Hf + (size_t)n * (NH * 32) + h * 32;
  float s = 0.f, d = 0.f;
#pragma unroll
  for (int c = 0; c < 32; ++c) {
    float v = row[c];
    s += v * wS[h * 32 + c];
    d += v * wD[h * 32 + c];
  }
  aS[t] = s;
  aD[t] = d;
}

template<int NH>
__global__ void k_edge_max(const int* __restrict__ ei, int E, int Nn,
                           const float* __restrict__ aS, const float* __restrict__ aD,
                           unsigned* __restrict__ menc) {
  int total = (E + Nn) * NH;
  for (int t = blockIdx.x * blockDim.x + threadIdx.x; t < total;
       t += gridDim.x * blockDim.x) {
    int e = t / NH, h = t - e * NH;
    int s, d; get_edge(ei, E, e, s, d);
    float v = lrelu02(aS[s * NH + h] + aD[d * NH + h]);
    atomicMax(&menc[d * NH + h], fenc(v));
  }
}

template<int NH>
__global__ void k_edge_sum(const int* __restrict__ ei, int E, int Nn,
                           const float* __restrict__ aS, const float* __restrict__ aD,
                           const unsigned* __restrict__ menc, float* __restrict__ ssum) {
  int total = (E + Nn) * NH;
  for (int t = blockIdx.x * blockDim.x + threadIdx.x; t < total;
       t += gridDim.x * blockDim.x) {
    int e = t / NH, h = t - e * NH;
    int s, d; get_edge(ei, E, e, s, d);
    float v = lrelu02(aS[s * NH + h] + aD[d * NH + h]);
    atomicAdd(&ssum[d * NH + h], __expf(v - fdec(menc[d * NH + h])));
  }
}

// one wave per edge; lane covers NH*32/32 consecutive channels
template<int NH>
__global__ void k_edge_agg(const int* __restrict__ ei, int E, int Nn,
                           const float* __restrict__ aS, const float* __restrict__ aD,
                           const unsigned* __restrict__ menc, const float* __restrict__ ssum,
                           const float* __restrict__ Hp, float* __restrict__ agg) {
  const int COLS = NH * 32;
  const int PER  = COLS / 32;
  int e = blockIdx.x * blockDim.y + threadIdx.y;
  if (e >= E + Nn) return;
  int lane = threadIdx.x;
  int s, d; get_edge(ei, E, e, s, d);
  int col0 = lane * PER;
  int h = col0 >> 5;
  float v = lrelu02(aS[s * NH + h] + aD[d * NH + h]);
  float alpha = __expf(v - fdec(menc[d * NH + h])) / ssum[d * NH + h];
  const float* hsrc = Hp  + (size_t)s * COLS + col0;
  float*       adst = agg + (size_t)d * COLS + col0;
  if (PER == 4) {
    float4 hv = *(const float4*)hsrc;   // 16B-aligned vector gather
    atomicAdd(adst + 0, alpha * hv.x);
    atomicAdd(adst + 1, alpha * hv.y);
    atomicAdd(adst + 2, alpha * hv.z);
    atomicAdd(adst + 3, alpha * hv.w);
  } else {
#pragma unroll
    for (int i = 0; i < PER; ++i)
      atomicAdd(adst + i, alpha * hsrc[i]);
  }
}

__global__ void k_bias_elu(const float* __restrict__ agg, const float* __restrict__ b,
                           float* __restrict__ out, int Nn, int D) {
  int t = blockIdx.x * blockDim.x + threadIdx.x;
  if (t >= Nn * D) return;
  out[t] = eluf(agg[t] + b[t & (D - 1)]);
}

__global__ void k_bias_elu_res(const float* __restrict__ agg, const float* __restrict__ b,
                               float* hio, int Nn, int D) {
  int t = blockIdx.x * blockDim.x + threadIdx.x;
  if (t >= Nn * D) return;
  hio[t] = eluf(agg[t] + b[t & (D - 1)]) + hio[t];
}

__global__ void k_head(const float* __restrict__ h3,
                       const float* __restrict__ rw, const float* __restrict__ rb,
                       const float* __restrict__ cw, const float* __restrict__ cb,
                       float* __restrict__ out, int Nn) {
  int n = blockIdx.x * blockDim.x + threadIdx.x;
  if (n >= Nn) return;
  const float* r = h3 + (size_t)n * 32;
  float a = 0.f, c = 0.f;
#pragma unroll
  for (int i = 0; i < 32; ++i) {
    float v = r[i];
    a += v * rw[i];
    c += v * cw[i];
  }
  out[n]      = a + rb[0];
  out[Nn + n] = c + cb[0];
}

extern "C" void kernel_launch(void* const* d_in, const int* in_sizes, int n_in,
                              void* d_out, int out_size, void* d_ws, size_t ws_size,
                              hipStream_t stream) {
  (void)n_in; (void)out_size; (void)ws_size;
  const float* x   = (const float*)d_in[0];
  const int*   ei  = (const int*)  d_in[1];
  const float* W1  = (const float*)d_in[2];
  const float* aS1 = (const float*)d_in[3];
  const float* aD1 = (const float*)d_in[4];
  const float* b1  = (const float*)d_in[5];
  const float* W2  = (const float*)d_in[6];
  const float* aS2 = (const float*)d_in[7];
  const float* aD2 = (const float*)d_in[8];
  const float* b2  = (const float*)d_in[9];
  const float* W3  = (const float*)d_in[10];
  const float* aS3 = (const float*)d_in[11];
  const float* aD3 = (const float*)d_in[12];
  const float* b3  = (const float*)d_in[13];
  const float* rw  = (const float*)d_in[14];
  const float* rb  = (const float*)d_in[15];
  const float* cw  = (const float*)d_in[16];
  const float* cb  = (const float*)d_in[17];

  const int Nn   = in_sizes[0] / 16;   // 100000, divisible by 16
  const int E    = in_sizes[1] / 2;    // 1600000
  const int Etot = E + Nn;

  // workspace layout (fp32): 3x [N,128] + per-node scalars + packed f16 weights
  float* bufA = (float*)d_ws;                         // running node features h
  float* bufB = bufA + (size_t)Nn * 128;              // projected features hp
  float* bufC = bufB + (size_t)Nn * 128;              // aggregation accumulator
  float* aSb  = bufC + (size_t)Nn * 128;              // [N,4]
  float* aDb  = aSb + (size_t)Nn * 4;                 // [N,4]
  unsigned* menc = (unsigned*)(aDb + (size_t)Nn * 4); // [N,4] encoded seg-max
  float* ssum = (float*)(menc + (size_t)Nn * 4);      // [N,4] seg-sum
  _Float16* Bp1 = (_Float16*)(ssum + (size_t)Nn * 4); // 1*128*32 halfs
  _Float16* Bp2 = Bp1 + 1 * 128 * 32;                 // 4*128*32 halfs
  _Float16* Bp3 = Bp2 + 4 * 128 * 32;                 // 4*32*32 halfs

  auto cdiv = [](int a, int b) { return (a + b - 1) / b; };
  const int ggrid = Nn / 16;
  const dim3 aggb(32, 8);
  const int agggrid = cdiv(Etot, 8);

  // pre-pack weights into f16 B-fragment order (tiny)
  k_packB<16, 128><<<cdiv(1 * 128 * 32, 256), 256, 0, stream>>>(W1, Bp1);
  k_packB<128, 128><<<cdiv(4 * 128 * 32, 256), 256, 0, stream>>>(W2, Bp2);
  k_packB<128, 32><<<cdiv(4 * 32 * 32, 256), 256, 0, stream>>>(W3, Bp3);

  // ================= Layer 1 (K=16 -> 128, 4 heads) =================
  k_gemm_wmma<16, 128><<<ggrid, 256, 0, stream>>>(x, Bp1, bufB);
  k_attn<4><<<cdiv(Nn * 4, 256), 256, 0, stream>>>(bufB, aS1, aD1, aSb, aDb, Nn);
  k_zero<<<4096, 256, 0, stream>>>(bufC, (size_t)Nn * 128);
  k_zero<<<1024, 256, 0, stream>>>((float*)menc, (size_t)Nn * 8);
  k_edge_max<4><<<cdiv(Etot * 4, 256), 256, 0, stream>>>(ei, E, Nn, aSb, aDb, menc);
  k_edge_sum<4><<<cdiv(Etot * 4, 256), 256, 0, stream>>>(ei, E, Nn, aSb, aDb, menc, ssum);
  k_edge_agg<4><<<agggrid, aggb, 0, stream>>>(ei, E, Nn, aSb, aDb, menc, ssum, bufB, bufC);
  k_bias_elu<<<cdiv(Nn * 128, 256), 256, 0, stream>>>(bufC, b1, bufA, Nn, 128);

  // ================= Layer 2 (128 -> 128, 4 heads, residual) ========
  k_gemm_wmma<128, 128><<<ggrid, 256, 0, stream>>>(bufA, Bp2, bufB);
  k_attn<4><<<cdiv(Nn * 4, 256), 256, 0, stream>>>(bufB, aS2, aD2, aSb, aDb, Nn);
  k_zero<<<4096, 256, 0, stream>>>(bufC, (size_t)Nn * 128);
  k_zero<<<1024, 256, 0, stream>>>((float*)menc, (size_t)Nn * 8);
  k_edge_max<4><<<cdiv(Etot * 4, 256), 256, 0, stream>>>(ei, E, Nn, aSb, aDb, menc);
  k_edge_sum<4><<<cdiv(Etot * 4, 256), 256, 0, stream>>>(ei, E, Nn, aSb, aDb, menc, ssum);
  k_edge_agg<4><<<agggrid, aggb, 0, stream>>>(ei, E, Nn, aSb, aDb, menc, ssum, bufB, bufC);
  k_bias_elu_res<<<cdiv(Nn * 128, 256), 256, 0, stream>>>(bufC, b2, bufA, Nn, 128);

  // ================= Layer 3 (128 -> 32, 1 head, mean==identity) ====
  k_gemm_wmma<128, 32><<<ggrid, 64, 0, stream>>>(bufA, Bp3, bufB);
  k_attn<1><<<cdiv(Nn, 256), 256, 0, stream>>>(bufB, aS3, aD3, aSb, aDb, Nn);
  k_zero<<<4096, 256, 0, stream>>>(bufC, (size_t)Nn * 32);
  k_zero<<<1024, 256, 0, stream>>>((float*)menc, (size_t)Nn * 8);
  k_edge_max<1><<<cdiv(Etot, 256), 256, 0, stream>>>(ei, E, Nn, aSb, aDb, menc);
  k_edge_sum<1><<<cdiv(Etot, 256), 256, 0, stream>>>(ei, E, Nn, aSb, aDb, menc, ssum);
  k_edge_agg<1><<<agggrid, aggb, 0, stream>>>(ei, E, Nn, aSb, aDb, menc, ssum, bufB, bufC);
  // h3 = elu(agg + b3) -> bufB (hp3 is dead)
  k_bias_elu<<<cdiv(Nn * 32, 256), 256, 0, stream>>>(bufC, b3, bufB, Nn, 32);

  // ================= Heads =================
  k_head<<<cdiv(Nn, 256), 256, 0, stream>>>(bufB, rw, rb, cw, cb, (float*)d_out, Nn);
}
